// MemoryLayer_13134009991919
// MI455X (gfx1250) — matrix operand order repaired and checked by reading, not compiled, not run on hardware
//
#include <hip/hip_runtime.h>

#define B_   64
#define S_   256
#define I_   256
#define H2   1024   // 2*H
#define HH   512    // Hh
#define MM   (B_*S_)   // 16384 GEMM rows
#define KK   I_        // 256
#define NN   H2        // 1024
#define ROWS 32        // mem rows per scan block

typedef __attribute__((ext_vector_type(16))) __bf16 bf16x16;
typedef __attribute__((ext_vector_type(8)))  float  f32x8;

__device__ __forceinline__ float fast_tanh(float x) {
    // tanh(x) = (1 - e^{-2x}) / (1 + e^{-2x});  e^{-2x} = 2^{-2*log2(e)*x}
    float e = __builtin_amdgcn_exp2f(-2.8853900817779268f * x);
    return (1.0f - e) * __builtin_amdgcn_rcpf(1.0f + e);
}

__device__ __forceinline__ float4 upd4(float4 m, float um, float uk, float4 kt) {
    float4 r;
    r.x = fmaf(m.x, um, uk * kt.x);
    r.y = fmaf(m.y, um, uk * kt.y);
    r.z = fmaf(m.z, um, uk * kt.z);
    r.w = fmaf(m.w, um, uk * kt.w);
    return r;
}
__device__ __forceinline__ float dot4(float4 a, float4 b) {
    return a.x*b.x + a.y*b.y + a.z*b.z + a.w*b.w;
}

// ---------------- f32 -> bf16 conversion ----------------
__global__ __launch_bounds__(256)
void cvt_bf16_kernel(const float* __restrict__ src, __bf16* __restrict__ dst, int n) {
    int i4 = (blockIdx.x * 256 + threadIdx.x) * 4;
    if (i4 + 3 < n) {
        float4 f = *(const float4*)(src + i4);
        union { __bf16 h[4]; uint2 u; } p;
        p.h[0] = (__bf16)f.x; p.h[1] = (__bf16)f.y;
        p.h[2] = (__bf16)f.z; p.h[3] = (__bf16)f.w;
        *(uint2*)(dst + i4) = p.u;
    }
}

// ---------------- WMMA GEMM: C[M,N] = A[M,K] * W[N,K]^T ----------------
// One wave computes a 16x64 output strip (4 accumulators), K-loop step 32.
__global__ __launch_bounds__(256)
void gemm_bf16_wmma_kernel(const __bf16* __restrict__ A, const __bf16* __restrict__ Wb,
                           float* __restrict__ C) {
    const int lane = threadIdx.x & 31;
    const int wave = threadIdx.x >> 5;
    const int gw   = blockIdx.x * 8 + wave;      // global wave id
    const int m0   = (gw >> 4) * 16;             // 1024 M-tiles
    const int n0   = (gw & 15) * 64;             // 16 N-groups of 64
    const int g    = lane >> 4;                  // half-wave group
    const int l    = lane & 15;

    f32x8 acc[4] = {};
    const __bf16* arow = A + (size_t)(m0 + l) * KK;

    for (int kk = 0; kk < KK; kk += 32) {
        // A 16x32 bf16 frag: lanes 0-15 hold K {0..7,16..23}, lanes 16-31 K {8..15,24..31}
        union { bf16x16 v; uint4 u[2]; } a;
        a.u[0] = *(const uint4*)(arow + kk + g * 8);
        a.u[1] = *(const uint4*)(arow + kk + 16 + g * 8);
#pragma unroll
        for (int jt = 0; jt < 4; ++jt) {
            // B 32x16 bf16 frag: lane holds column n = n0+jt*16+l, K {g*8..g*8+7, 16+g*8..16+g*8+7}
            const __bf16* brow = Wb + (size_t)(n0 + jt * 16 + l) * KK + kk + g * 8;
            union { bf16x16 v; uint4 u[2]; } b;
            b.u[0] = *(const uint4*)(brow);
            b.u[1] = *(const uint4*)(brow + 16);
            acc[jt] = __builtin_amdgcn_wmma_f32_16x16x32_bf16(
                false, a.v, false, b.v, (short)0, acc[jt], false, false);
        }
    }
    // D layout: VGPR j holds row m0 + j + 8*g, col n
#pragma unroll
    for (int jt = 0; jt < 4; ++jt) {
        const int ncol = n0 + jt * 16 + l;
#pragma unroll
        for (int j = 0; j < 8; ++j)
            C[(size_t)(m0 + j + 8 * g) * NN + ncol] = acc[jt][j];
    }
}

// ---------------- recurrent scan ----------------
// grid = B * (HH/ROWS) blocks; block b,rg owns mem rows [rg*ROWS, rg*ROWS+ROWS)
// of batch b. ks/kt (512) maintained redundantly per block (mem-independent).
__global__ __launch_bounds__(256)
void scan_kernel(const float* __restrict__ ibuf, float* __restrict__ mem,
                 float* __restrict__ keys, float* __restrict__ vals) {
    __shared__ __align__(16) float sKs[HH];
    __shared__ __align__(16) float sK [HH];
    __shared__ __align__(16) float sKt[HH];
    __shared__ float sVs[ROWS], sVt[ROWS], sIkv[ROWS];

    const int tid  = threadIdx.x;
    const int lane = tid & 31;
    const int wave = tid >> 5;
    const int b    = blockIdx.x >> 4;
    const int rg   = blockIdx.x & 15;
    const int r0   = rg * ROWS;

    const float alpha = 0.9048374180359595f;   // exp(-1/10)
    const float dtr   = 0.9512294245007140f;   // exp(-1/20)
    const float odtr  = 1.0f - dtr;
    const float LRc   = 0.01f;

    sKs[tid] = 0.0f; sKs[tid + 256] = 0.0f;
    sKt[tid] = 0.0f; sKt[tid + 256] = 0.0f;
    if (tid < ROWS) { sVs[tid] = 0.0f; sVt[tid] = 0.0f; }
    __syncthreads();

    float* memBase = mem + (size_t)b * HH * HH + (size_t)r0 * HH;
    const float* iBatch = ibuf + (size_t)b * S_ * H2;
    const int c0 = lane * 16;   // this lane's 16-column chunk

    for (int t = 0; t < S_; ++t) {
        const float* iRow = iBatch + (size_t)t * H2;
        // (a) ks update + k = tanh(ks)
        {
            float a0 = fmaf(alpha, sKs[tid],       iRow[tid]);
            float a1 = fmaf(alpha, sKs[tid + 256], iRow[tid + 256]);
            sKs[tid] = a0; sKs[tid + 256] = a1;
            sK [tid] = fast_tanh(a0); sK[tid + 256] = fast_tanh(a1);
        }
        __syncthreads();
        // (b) fused: apply step t-1 outer-product update, dot with k_t
        if (t > 0) {
            float4 kr0 = *(const float4*)(sK  + c0);
            float4 kr1 = *(const float4*)(sK  + c0 + 4);
            float4 kr2 = *(const float4*)(sK  + c0 + 8);
            float4 kr3 = *(const float4*)(sK  + c0 + 12);
            float4 kt0 = *(const float4*)(sKt + c0);
            float4 kt1 = *(const float4*)(sKt + c0 + 4);
            float4 kt2 = *(const float4*)(sKt + c0 + 8);
            float4 kt3 = *(const float4*)(sKt + c0 + 12);
#pragma unroll
            for (int rr = 0; rr < 4; ++rr) {
                const int r = wave * 4 + rr;
                const float vtp = sVt[r];
                const float um  = 1.0f - LRc * vtp * vtp;
                const float uk  = LRc * vtp;
                float* mrow = memBase + (size_t)r * HH + c0;
                if (rr < 3) __builtin_prefetch(mrow + HH, 0, 1);
                float4 m0, m1, m2, m3;
                if (t >= 2) {
                    m0 = *(const float4*)(mrow);
                    m1 = *(const float4*)(mrow + 4);
                    m2 = *(const float4*)(mrow + 8);
                    m3 = *(const float4*)(mrow + 12);
                } else {
                    m0 = m1 = m2 = m3 = make_float4(0.f, 0.f, 0.f, 0.f);
                }
                m0 = upd4(m0, um, uk, kt0);
                m1 = upd4(m1, um, uk, kt1);
                m2 = upd4(m2, um, uk, kt2);
                m3 = upd4(m3, um, uk, kt3);
                *(float4*)(mrow)      = m0;
                *(float4*)(mrow + 4)  = m1;
                *(float4*)(mrow + 8)  = m2;
                *(float4*)(mrow + 12) = m3;
                float dot = dot4(m0, kr0) + dot4(m1, kr1)
                          + dot4(m2, kr2) + dot4(m3, kr3);
#pragma unroll
                for (int off = 16; off; off >>= 1)
                    dot += __shfl_xor(dot, off, 32);
                if (lane == 0) sIkv[r] = 0.2f * dot;
            }
        } else if (tid < ROWS) {
            sIkv[tid] = 0.0f;
        }
        __syncthreads();
        // (c) vs/v/vt for owned rows; kt update; emit keys/vals
        if (tid < ROWS) {
            const int r = tid;
            float vsv = fmaf(alpha, sVs[r], iRow[512 + r0 + r] + sIkv[r]);
            float v   = fast_tanh(vsv);
            sVs[r] = vsv;
            sVt[r] = fmaf(dtr, sVt[r], odtr * v);
            vals[(size_t)b * S_ * HH + (size_t)t * HH + r0 + r] = v;
        }
        {
            float k0 = sK[tid], k1 = sK[tid + 256];
            sKt[tid]       = fmaf(dtr, sKt[tid],       odtr * k0);
            sKt[tid + 256] = fmaf(dtr, sKt[tid + 256], odtr * k1);
            if (rg == 0) {
                float* kout = keys + (size_t)b * S_ * HH + (size_t)t * HH;
                kout[tid] = k0; kout[tid + 256] = k1;
            }
        }
        __syncthreads();
    }
    // final: apply step S-1 update to mem
    {
        float4 kt0 = *(const float4*)(sKt + c0);
        float4 kt1 = *(const float4*)(sKt + c0 + 4);
        float4 kt2 = *(const float4*)(sKt + c0 + 8);
        float4 kt3 = *(const float4*)(sKt + c0 + 12);
#pragma unroll
        for (int rr = 0; rr < 4; ++rr) {
            const int r = wave * 4 + rr;
            const float vtp = sVt[r];
            const float um  = 1.0f - LRc * vtp * vtp;
            const float uk  = LRc * vtp;
            float* mrow = memBase + (size_t)r * HH + c0;
            float4 m0 = *(const float4*)(mrow);
            float4 m1 = *(const float4*)(mrow + 4);
            float4 m2 = *(const float4*)(mrow + 8);
            float4 m3 = *(const float4*)(mrow + 12);
            *(float4*)(mrow)      = upd4(m0, um, uk, kt0);
            *(float4*)(mrow + 4)  = upd4(m1, um, uk, kt1);
            *(float4*)(mrow + 8)  = upd4(m2, um, uk, kt2);
            *(float4*)(mrow + 12) = upd4(m3, um, uk, kt3);
        }
    }
}

extern "C" void kernel_launch(void* const* d_in, const int* in_sizes, int n_in,
                              void* d_out, int out_size, void* d_ws, size_t ws_size,
                              hipStream_t stream) {
    const float* x = (const float*)d_in[0];   // (B, S, I)
    const float* W = (const float*)d_in[1];   // (2H, I)

    char* ws = (char*)d_ws;
    __bf16* xb   = (__bf16*)ws;                                     // 8 MiB
    __bf16* Wb   = (__bf16*)(ws + (size_t)MM * KK * 2);             // 512 KiB
    float*  ibuf = (float*)(ws + (size_t)MM * KK * 2
                               + (size_t)NN * KK * 2);              // 64 MiB

    float* memO = (float*)d_out;                                    // (B,HH,HH)
    float* keys = memO + (size_t)B_ * HH * HH;                      // (B,S,HH)
    float* vals = keys + (size_t)B_ * S_ * HH;                      // (B,S,HH)

    {
        int n = MM * KK;
        cvt_bf16_kernel<<<(n / 4 + 255) / 256, 256, 0, stream>>>(x, xb, n);
    }
    {
        int n = NN * KK;
        cvt_bf16_kernel<<<(n / 4 + 255) / 256, 256, 0, stream>>>(W, Wb, n);
    }
    // 16384 waves = (M/16)*(N/64); 8 waves per block
    gemm_bf16_wmma_kernel<<<2048, 256, 0, stream>>>(xb, Wb, ibuf);
    // 64 batches x 16 row-groups
    scan_kernel<<<B_ * (HH / ROWS), 256, 0, stream>>>(ibuf, memO, keys, vals);
}
